// LePEAttention_19250043420892
// MI455X (gfx1250) — matrix operand order repaired and checked by reading, not compiled, not run on hardware
//
#include <hip/hip_runtime.h>
#include <hip/hip_bf16.h>

typedef __attribute__((ext_vector_type(16))) _Float16 v16h;
typedef __attribute__((ext_vector_type(8)))  float    v8f;

#define HW_H 56
#define HW_W 56
#define LTOT (HW_H * HW_W)      // 3136
#define CDIM 512
#define HD 64
#define WSP 7
#define SWIN 392                 // 56*7 tokens per window
#define SPAD 416                 // 13 tiles of 32
#define NKT 13                   // key tiles of 32
#define NQT 25                   // query tiles of 16 (ceil(392/16))
#define QSCALE 0.125f            // 64^-0.5

__device__ __forceinline__ float red_max16(float x) {
  x = fmaxf(x, __shfl_xor(x, 1, 32));
  x = fmaxf(x, __shfl_xor(x, 2, 32));
  x = fmaxf(x, __shfl_xor(x, 4, 32));
  x = fmaxf(x, __shfl_xor(x, 8, 32));
  return x;
}

#define WMMA_F16(A, B, C) \
  __builtin_amdgcn_wmma_f32_16x16x32_f16(false, (A), false, (B), (short)0, (C), false, false)

__global__ __launch_bounds__(256) void lepe_attn_kernel(
    const float* __restrict__ qkv,     // (3, B, L, C)
    const float* __restrict__ conv_w,  // (C, 1, 3, 3)
    const float* __restrict__ conv_b,  // (C,)
    float* __restrict__ out)           // (B, L, C)
{
  // K staged row-major [SPAD][HD], V staged transposed [HD][SPAD], both f16.
  __shared__ alignas(32) _Float16 Kl[SPAD * HD];      // 52 KB
  __shared__ alignas(32) _Float16 Vt[HD * SPAD];      // 52 KB
  __shared__ alignas(32) _Float16 Pl[8][16 * 32];     // 8 KB per-wave P bounce

  const int bw   = blockIdx.x >> 3;   // window id: 0..63
  const int head = blockIdx.x & 7;
  const int b    = bw >> 3;           // batch
  const int wi   = bw & 7;            // window column group

  const size_t PLANE = (size_t)8 * LTOT * CDIM;
  const float* qp = qkv;
  const float* kp = qkv + PLANE;
  const float* vp = qkv + 2 * PLANE;

  const int tid = threadIdx.x;

  // ---- cooperative stage of K and V^T into LDS (f16) ----
  for (int it = tid; it < SPAD * 16; it += 256) {
    const int s  = it >> 4;
    const int d4 = (it & 15) << 2;
    float4 kv = make_float4(0.f, 0.f, 0.f, 0.f);
    float4 vv = make_float4(0.f, 0.f, 0.f, 0.f);
    if (s < SWIN) {
      const int li = (s / WSP) * HW_W + wi * WSP + (s % WSP);
      const size_t off = ((size_t)b * LTOT + li) * CDIM + head * HD + d4;
      kv = *(const float4*)(kp + off);
      vv = *(const float4*)(vp + off);
    }
    _Float16* kd = &Kl[s * HD + d4];
    kd[0] = (_Float16)kv.x; kd[1] = (_Float16)kv.y;
    kd[2] = (_Float16)kv.z; kd[3] = (_Float16)kv.w;
    Vt[(size_t)(d4 + 0) * SPAD + s] = (_Float16)vv.x;
    Vt[(size_t)(d4 + 1) * SPAD + s] = (_Float16)vv.y;
    Vt[(size_t)(d4 + 2) * SPAD + s] = (_Float16)vv.z;
    Vt[(size_t)(d4 + 3) * SPAD + s] = (_Float16)vv.w;
  }
  __syncthreads();

  const int wave  = tid >> 5;
  const int lane  = tid & 31;
  const int lhalf = lane >> 4;   // 0 or 1
  const int lm    = lane & 15;

  // B tile of ones in column 0: B[k][n] = (n==0) ? 1 : 0. Layout: element i
  // holds K=lhalf*16+i, N=lm -> every element is (lm==0). Pure register constant.
  v16h onesB;
  {
    const _Float16 ov = (lm == 0) ? (_Float16)1.0f : (_Float16)0.0f;
    #pragma unroll
    for (int i = 0; i < 16; ++i) onesB[i] = ov;
  }

  for (int qt = wave; qt < NQT; qt += 8) {
    const int s0 = qt * 16;

    // ---- load Q tile (16x64) into two A-layout v16h regs, scaled, f16 ----
    v16h qA[2];
    {
      const int sq = s0 + lm;
      #pragma unroll
      for (int c2 = 0; c2 < 2; ++c2) {
        v16h qa = {};
        if (sq < SWIN) {
          const int li = (sq / WSP) * HW_W + wi * WSP + (sq % WSP);
          const float* qb = qp + ((size_t)b * LTOT + li) * CDIM + head * HD
                               + c2 * 32 + lhalf * 16;
          #pragma unroll
          for (int g = 0; g < 2; ++g) {
            float4 f0 = *(const float4*)(qb + g * 8);
            float4 f1 = *(const float4*)(qb + g * 8 + 4);
            qa[g * 8 + 0] = (_Float16)(f0.x * QSCALE);
            qa[g * 8 + 1] = (_Float16)(f0.y * QSCALE);
            qa[g * 8 + 2] = (_Float16)(f0.z * QSCALE);
            qa[g * 8 + 3] = (_Float16)(f0.w * QSCALE);
            qa[g * 8 + 4] = (_Float16)(f1.x * QSCALE);
            qa[g * 8 + 5] = (_Float16)(f1.y * QSCALE);
            qa[g * 8 + 6] = (_Float16)(f1.z * QSCALE);
            qa[g * 8 + 7] = (_Float16)(f1.w * QSCALE);
          }
        }
        qA[c2] = qa;
      }
    }

    float mrow[8];
    v8f acc[4];
    v8f accS = (v8f){};   // row sums of P via ones-column WMMA
    #pragma unroll
    for (int r = 0; r < 8; ++r) mrow[r] = -1e30f;
    #pragma unroll
    for (int t4 = 0; t4 < 4; ++t4) acc[t4] = (v8f){};

    // ---- flash loop over 13 key tiles of 32 ----
    for (int kt = 0; kt < NKT; ++kt) {
      const int kb = kt * 32;

      // scores: 16x32 = two 16x16 D tiles, contraction over HD=64
      v8f sc[2];
      #pragma unroll
      for (int t = 0; t < 2; ++t) {
        v8f s8 = (v8f){};
        const v16h* bk0 = (const v16h*)&Kl[(size_t)(kb + t * 16 + lm) * HD + lhalf * 16];
        const v16h* bk1 = (const v16h*)&Kl[(size_t)(kb + t * 16 + lm) * HD + 32 + lhalf * 16];
        s8 = WMMA_F16(qA[0], *bk0, s8);
        s8 = WMMA_F16(qA[1], *bk1, s8);
        sc[t] = s8;
      }

      // mask padded key columns
      #pragma unroll
      for (int t = 0; t < 2; ++t) {
        if (kb + t * 16 + lm >= SWIN) {
          #pragma unroll
          for (int r = 0; r < 8; ++r) sc[t][r] = -1e30f;
        }
      }

      // online softmax max update (row-wise; lane holds rows r+8*lhalf, col lm)
      float pv[2][8], corr[8];
      #pragma unroll
      for (int r = 0; r < 8; ++r) {
        const float tm = red_max16(fmaxf(sc[0][r], sc[1][r]));
        const float nm = fmaxf(mrow[r], tm);
        corr[r] = __expf(mrow[r] - nm);
        mrow[r] = nm;
        pv[0][r] = __expf(sc[0][r] - nm);
        pv[1][r] = __expf(sc[1][r] - nm);
      }
      #pragma unroll
      for (int t4 = 0; t4 < 4; ++t4)
        #pragma unroll
        for (int r = 0; r < 8; ++r) acc[t4][r] *= corr[r];
      #pragma unroll
      for (int r = 0; r < 8; ++r) accS[r] *= corr[r];

      // bounce P through LDS: D-layout -> A-layout (wave-local, in-order DS)
      _Float16* pw = &Pl[wave][0];
      #pragma unroll
      for (int t = 0; t < 2; ++t)
        #pragma unroll
        for (int r = 0; r < 8; ++r)
          pw[(r + 8 * lhalf) * 32 + t * 16 + lm] = (_Float16)pv[t][r];

      v16h pA;
      {
        const _Float16* pr = &Pl[wave][lm * 32 + lhalf * 16];
        #pragma unroll
        for (int i = 0; i < 8; ++i) { pA[i] = pr[i]; pA[8 + i] = pr[8 + i]; }
      }

      // P(16x32) x V(32x64): 4 output tiles, one WMMA each (K=32)
      #pragma unroll
      for (int t4 = 0; t4 < 4; ++t4) {
        const v16h* bv =
            (const v16h*)&Vt[(size_t)(t4 * 16 + lm) * SPAD + kb + lhalf * 16];
        acc[t4] = WMMA_F16(pA, *bv, acc[t4]);
      }
      // row sums of P accumulate into column 0 of accS (register-only B)
      accS = WMMA_F16(pA, onesB, accS);
    }

    // ---- epilogue: normalize, add LePE depthwise 3x3 from Vt, scatter out ----
    float w9[4][9], bias4[4];
    #pragma unroll
    for (int t4 = 0; t4 < 4; ++t4) {
      const int c = head * HD + t4 * 16 + lm;
      bias4[t4] = conv_b[c];
      #pragma unroll
      for (int j = 0; j < 9; ++j) w9[t4][j] = conv_w[c * 9 + j];
    }

    #pragma unroll
    for (int r = 0; r < 8; ++r) {
      const int m = r + 8 * lhalf;
      const int s = s0 + m;
      // broadcast row-sum from column-0 lane of this half (lane 0 or 16)
      const float lsum = __shfl(accS[r], lhalf << 4, 32);
      if (s >= SWIN) continue;
      const int h  = s / WSP;
      const int wf = s % WSP;
      const int li = h * HW_W + wi * WSP + wf;
      const float invl = 1.0f / lsum;
      #pragma unroll
      for (int t4 = 0; t4 < 4; ++t4) {
        const int d = t4 * 16 + lm;
        float lepe = bias4[t4];
        #pragma unroll
        for (int dh = -1; dh <= 1; ++dh)
          #pragma unroll
          for (int dw = -1; dw <= 1; ++dw) {
            const int hh = h + dh, ww = wf + dw;
            if (hh >= 0 && hh < HW_H && ww >= 0 && ww < WSP)
              lepe += (float)Vt[(size_t)d * SPAD + hh * WSP + ww] *
                      w9[t4][(dh + 1) * 3 + (dw + 1)];
          }
        out[((size_t)b * LTOT + li) * CDIM + head * HD + d] =
            acc[t4][r] * invl + lepe;
      }
    }
  }
}

extern "C" void kernel_launch(void* const* d_in, const int* in_sizes, int n_in,
                              void* d_out, int out_size, void* d_ws, size_t ws_size,
                              hipStream_t stream) {
  (void)in_sizes; (void)n_in; (void)out_size; (void)d_ws; (void)ws_size;
  const float* qkv    = (const float*)d_in[0];
  const float* conv_w = (const float*)d_in[1];
  const float* conv_b = (const float*)d_in[2];
  float* out          = (float*)d_out;
  dim3 grid(512), block(256);
  lepe_attn_kernel<<<grid, block, 0, stream>>>(qkv, conv_w, conv_b, out);
}
